// TwoBranchASPM_PixelGate_33663953666138
// MI455X (gfx1250) — matrix operand order repaired and checked by reading, not compiled, not run on hardware
//
#include <hip/hip_runtime.h>
#include <hip/hip_bf16.h>
#include <math.h>

// ---------------- problem constants ----------------
#define BATCH 16
#define CH    64
#define HH    128
#define WW    128
#define HWSZ  (HH*WW)          // 16384
#define CG    32
#define HL    64               // low-res H
#define WL    64               // low-res W
#define PL    (HL*WL)          // 4096

typedef __attribute__((ext_vector_type(16))) _Float16 v16h;
typedef __attribute__((ext_vector_type(8)))  float    v8f;

// ================= m_low: channel mean + 2x2 avgpool =================
__global__ void k_mlow(const float* __restrict__ x, float* __restrict__ mlow) {
  int idx = blockIdx.x * 256 + threadIdx.x;          // BATCH*PL threads
  int wo = idx % WL, ho = (idx / WL) % HL, b = idx / PL;
  const float* xb = x + (size_t)b * CH * HWSZ;
  int p00 = (2*ho)*WW + 2*wo;
  float s = 0.f;
  for (int c = 0; c < CH; ++c) {
    const float* xs = xb + (size_t)c * HWSZ + p00;
    s += xs[0] + xs[1] + xs[WW] + xs[WW+1];
  }
  mlow[idx] = s * (1.0f / (CH * 4.0f));
}

// ========== WMMA im2col conv3x3 (pad effective 1): CIN -> 32 ==========
// One wave: 16 output pixels x 32 output channels; B-regs (im2col activations)
// reused across both 16-channel A tiles.  K loop fully unrolled so the
// k -> (ci,kh,kw) decode constant-folds into immediate address offsets.
template<int CIN, int INH, int STRIDE, bool RELU>
__global__ void k_conv3x3_wmma(const float* __restrict__ in, const float* __restrict__ w,
                               float* __restrict__ out) {
  constexpr int KTOT   = CIN * 9;
  constexpr int KSTEPS = (KTOT + 31) / 32;
  constexpr int INSZ   = INH * INH;
  int wave = threadIdx.x >> 5;
  int lane = threadIdx.x & 31;
  int tile = blockIdx.x * 8 + wave;                  // BATCH*PL/16 tiles
  int b  = tile >> 8;                                // PL/16 = 256 tiles per batch
  int p0 = (tile & 255) * 16;
  int half = lane >> 4, l = lane & 15;
  int p  = p0 + l;
  int wo = p & (WL - 1), ho = p >> 6;                // 16-pixel tile = one row chunk
  const float* inb = in + (size_t)b * CIN * INSZ;
  v8f acc0 = {}, acc1 = {};
  #pragma unroll
  for (int s = 0; s < KSTEPS; ++s) {
    // B regs: element j <-> K = 32s + 16*half + j (contiguous k order, same as A)
    v16h B;
    #pragma unroll
    for (int j = 0; j < 16; ++j) {
      int k = 32*s + 16*half + j;
      float v = 0.f;
      if (k < KTOT) {
        int ci = k / 9, r = k % 9, kh = r / 3, kw = r % 3;
        int h  = STRIDE*ho - 1 + kh;
        int w2 = STRIDE*wo - 1 + kw;
        if ((unsigned)h < (unsigned)INH && (unsigned)w2 < (unsigned)INH)
          v = inb[(size_t)ci * INSZ + h*INH + w2];
      }
      B[j] = (_Float16)v;
    }
    // A regs: element j <-> K = 32s + 16*half + j
    v16h A0, A1;
    #pragma unroll
    for (int j = 0; j < 16; ++j) {
      int k = 32*s + 16*half + j;
      float a0 = 0.f, a1 = 0.f;
      if (k < KTOT) {
        a0 = w[(size_t)l        * KTOT + k];
        a1 = w[(size_t)(16 + l) * KTOT + k];
      }
      A0[j] = (_Float16)a0;
      A1[j] = (_Float16)a1;
    }
    acc0 = __builtin_amdgcn_wmma_f32_16x16x32_f16(false, A0, false, B, (short)0, acc0, false, false);
    acc1 = __builtin_amdgcn_wmma_f32_16x16x32_f16(false, A1, false, B, (short)0, acc1, false, false);
  }
  // D layout: VGPR r <-> M = r + 8*half, lane&15 = N
  float* ob = out + (size_t)b * 32 * PL + p;
  #pragma unroll
  for (int r = 0; r < 8; ++r) {
    int m = r + 8*half;
    float v0 = acc0[r], v1 = acc1[r];
    if (RELU) { v0 = fmaxf(v0, 0.f); v1 = fmaxf(v1, 0.f); }
    ob[(size_t)m        * PL] = v0;
    ob[(size_t)(16 + m) * PL] = v1;
  }
}

// ============ GroupNorm stats -> per-channel scale/bias  ============
// scale[b*C+c] = gn_w*rstd ; bias[b*C+c] = gn_b - mu*rstd*gn_w
__global__ void k_gnstats(const float* __restrict__ data,
                          const float* __restrict__ gnw, const float* __restrict__ gnb,
                          float* __restrict__ scale, float* __restrict__ bias,
                          int chPerGroup, int spatial, int C) {
  int numGroups = C / chPerGroup;
  int g  = blockIdx.x;                     // b*numGroups + gi
  int b  = g / numGroups, gi = g % numGroups;
  const float* base = data + ((size_t)b * C + (size_t)gi * chPerGroup) * spatial;
  int n = chPerGroup * spatial;
  float s = 0.f, s2 = 0.f;
  for (int i = threadIdx.x; i < n; i += 256) {
    float v = base[i];
    s += v; s2 += v * v;
  }
  __shared__ float r0[256], r1[256];
  r0[threadIdx.x] = s; r1[threadIdx.x] = s2;
  __syncthreads();
  for (int off = 128; off > 0; off >>= 1) {
    if (threadIdx.x < off) { r0[threadIdx.x] += r0[threadIdx.x+off];
                             r1[threadIdx.x] += r1[threadIdx.x+off]; }
    __syncthreads();
  }
  if (threadIdx.x < (unsigned)chPerGroup) {
    float inv = 1.0f / (float)n;
    float mu  = r0[0] * inv;
    float var = r1[0] * inv - mu * mu;
    float rstd = rsqrtf(var + 1e-5f);
    int c = gi * chPerGroup + threadIdx.x;
    float wv = gnw[c], bv = gnb[c];
    scale[b*C + c] = wv * rstd;
    bias [b*C + c] = bv - mu * rstd * wv;
  }
}

// ===== build 34-ch gate input: GN+ReLU(g1) | sobel edge | local var =====
__global__ void k_g1cat(const float* __restrict__ g1raw,
                        const float* __restrict__ sg, const float* __restrict__ bg,
                        const float* __restrict__ mlow, float* __restrict__ cat) {
  int idx = blockIdx.x * 256 + threadIdx.x;          // BATCH*34*PL
  int p = idx % PL, c = (idx / PL) % 34, b = idx / (PL * 34);
  float out;
  if (c < 32) {
    float v = g1raw[((size_t)b * CG + c) * PL + p];
    out = fmaxf(sg[b*CG + c] * v + bg[b*CG + c], 0.f);
  } else {
    int wo = p % WL, ho = p / WL;
    const float* m = mlow + (size_t)b * PL;
    if (c == 32) {                                   // sobel magnitude
      const float KX[9] = {-1,0,1,-2,0,2,-1,0,1};
      const float KY[9] = {-1,-2,-1,0,0,0,1,2,1};
      float gx = 0.f, gy = 0.f;
      #pragma unroll
      for (int i = 0; i < 3; ++i)
        #pragma unroll
        for (int j = 0; j < 3; ++j) {
          int h = ho + i - 1, w2 = wo + j - 1;
          if ((unsigned)h < (unsigned)HL && (unsigned)w2 < (unsigned)WL) {
            float v = m[h*WL + w2];
            gx += KX[i*3+j] * v; gy += KY[i*3+j] * v;
          }
        }
      out = sqrtf(gx*gx + gy*gy + 1e-6f);
    } else {                                         // 5x5 local variance
      float s = 0.f, s2 = 0.f;
      #pragma unroll
      for (int i = -2; i <= 2; ++i)
        #pragma unroll
        for (int j = -2; j <= 2; ++j) {
          int h = ho + i, w2 = wo + j;
          if ((unsigned)h < (unsigned)HL && (unsigned)w2 < (unsigned)WL) {
            float v = m[h*WL + w2];
            s += v; s2 += v * v;
          }
        }
      float mean = s * (1.f/25.f), mean2 = s2 * (1.f/25.f);
      out = fmaxf(mean2 - mean*mean, 0.f);
    }
  }
  cat[idx] = out;
}

// ============ gate head: 1x1 conv 32 -> 2 + bias ============
__global__ void k_logits(const float* __restrict__ h, const float* __restrict__ w,
                         const float* __restrict__ bb, float* __restrict__ logits) {
  int idx = blockIdx.x * 256 + threadIdx.x;          // BATCH*PL
  int p = idx % PL, b = idx / PL;
  const float* hb = h + (size_t)b * CG * PL + p;
  float a0 = bb[0], a1 = bb[1];
  for (int ci = 0; ci < CG; ++ci) {
    float v = hb[(size_t)ci * PL];
    a0 += w[ci]      * v;
    a1 += w[CG + ci] * v;
  }
  float* lb = logits + (size_t)b * 2 * PL;
  lb[p]      = a0;
  lb[PL + p] = a1;
}

// == fused: dual depthwise + bilinear-upsampled softmax gate -> y ==
__global__ void k_fuse(const float* __restrict__ x, const float* __restrict__ w3,
                       const float* __restrict__ w5, const float* __restrict__ logits,
                       float* __restrict__ y) {
  size_t idx = (size_t)blockIdx.x * 256 + threadIdx.x;   // BATCH*CH*HWSZ
  int wp = idx % WW, hp = (idx / WW) % HH;
  int c  = (idx / HWSZ) % CH; int b = idx / ((size_t)HWSZ * CH);
  const float* xs = x + ((size_t)b * CH + c) * HWSZ;
  // 3x3 depthwise
  float ys = 0.f;
  {
    const float* wp3 = w3 + (size_t)c * 9;
    #pragma unroll
    for (int i = 0; i < 3; ++i) {
      int h = hp + i - 1;
      if ((unsigned)h >= (unsigned)HH) continue;
      #pragma unroll
      for (int j = 0; j < 3; ++j) {
        int w2 = wp + j - 1;
        if ((unsigned)w2 >= (unsigned)WW) continue;
        ys += wp3[i*3+j] * xs[h*WW + w2];
      }
    }
  }
  // 5x5 dilation-2 depthwise
  float yl = 0.f;
  {
    const float* wp5 = w5 + (size_t)c * 25;
    #pragma unroll
    for (int i = 0; i < 5; ++i) {
      int h = hp + 2*i - 4;
      if ((unsigned)h >= (unsigned)HH) continue;
      #pragma unroll
      for (int j = 0; j < 5; ++j) {
        int w2 = wp + 2*j - 4;
        if ((unsigned)w2 >= (unsigned)WW) continue;
        yl += wp5[i*5+j] * xs[h*WW + w2];
      }
    }
  }
  // bilinear x2 upsample of 2-ch logits (half-pixel centers, clamped)
  float sh = 0.5f * hp - 0.25f, sw = 0.5f * wp - 0.25f;
  int h0 = (int)floorf(sh), w0 = (int)floorf(sw);
  float fh = sh - h0, fw = sw - w0;
  int h0c = min(max(h0, 0), HL-1), h1c = min(max(h0+1, 0), HL-1);
  int w0c = min(max(w0, 0), WL-1), w1c = min(max(w0+1, 0), WL-1);
  const float* l0p = logits + (size_t)b * 2 * PL;
  const float* l1p = l0p + PL;
  float c00 = (1.f-fh)*(1.f-fw), c01 = (1.f-fh)*fw, c10 = fh*(1.f-fw), c11 = fh*fw;
  float l0 = c00*l0p[h0c*WL+w0c] + c01*l0p[h0c*WL+w1c] + c10*l0p[h1c*WL+w0c] + c11*l0p[h1c*WL+w1c];
  float l1 = c00*l1p[h0c*WL+w0c] + c01*l1p[h0c*WL+w1c] + c10*l1p[h1c*WL+w0c] + c11*l1p[h1c*WL+w1c];
  float g0 = 1.f / (1.f + expf(l1 - l0));            // softmax over 2
  y[idx] = g0 * ys + (1.f - g0) * yl;
}

// ======= WMMA pointwise 1x1 conv: out[co,p] = sum_ci W[co,ci]*relu(sA*y+bA) =======
// One wave -> 16 pixels x 64 output channels.  8x v_wmma_f32_16x16x32_f16 per wave.
__global__ void k_pwgemm(const float* __restrict__ y, const float* __restrict__ pw_w,
                         const float* __restrict__ sA, const float* __restrict__ bA,
                         float* __restrict__ pw_raw) {
  int wave = threadIdx.x >> 5;
  int lane = threadIdx.x & 31;
  int tile = blockIdx.x * 8 + wave;                  // 16384 pixel tiles total
  int b  = tile >> 10;                               // 1024 tiles per batch
  int p0 = (tile & 1023) * 16;
  int half = lane >> 4, l = lane & 15;
  int p = p0 + l;
  const float* yb = y + (size_t)b * CH * HWSZ;
  const float* sAb = sA + b * CH;
  const float* bAb = bA + b * CH;

  // B regs: K x N = 32 x 16 per step; element j <-> K = 16*half + j (contiguous k order, same as A)
  v16h B0, B1;
  #pragma unroll
  for (int j = 0; j < 16; ++j) {
    int k  = 16*half + j;
    float v0 = yb[(size_t)k        * HWSZ + p];
    float v1 = yb[(size_t)(32 + k) * HWSZ + p];
    v0 = fmaxf(sAb[k]      * v0 + bAb[k],      0.f);
    v1 = fmaxf(sAb[32 + k] * v1 + bAb[32 + k], 0.f);
    B0[j] = (_Float16)v0;
    B1[j] = (_Float16)v1;
  }

  float* outp = pw_raw + (size_t)b * CH * HWSZ + p;
  #pragma unroll
  for (int t = 0; t < 4; ++t) {                      // 4 output-channel tiles of 16
    int co = t * 16 + l;                             // A row = l for both halves
    v8f acc = {};
    #pragma unroll
    for (int s = 0; s < 2; ++s) {
      // A regs: M x K = 16 x 32; element j <-> K = 16*half + j
      v16h A;
      #pragma unroll
      for (int j = 0; j < 16; ++j) {
        int k = 16*half + j;
        A[j] = (_Float16)pw_w[(size_t)co * CH + 32*s + k];
      }
      acc = __builtin_amdgcn_wmma_f32_16x16x32_f16(
          false, A, false, (s == 0 ? B0 : B1), (short)0, acc, false, false);
    }
    // D layout: VGPR r <-> M = r + 8*half, lane&15 = N
    #pragma unroll
    for (int r = 0; r < 8; ++r) {
      int m = t * 16 + r + 8*half;
      outp[(size_t)m * HWSZ] = acc[r];
    }
  }
}

// ============ zero small accumulator buffer ============
__global__ void k_zero(float* __restrict__ p, int n) {
  int i = blockIdx.x * 256 + threadIdx.x;
  if (i < n) p[i] = 0.f;
}

// == residual: out = x + gamma*relu(sB*pw+bB); accumulate per-(b,c) sums ==
__global__ void k_residual(const float* __restrict__ x, const float* __restrict__ pw,
                           const float* __restrict__ sB, const float* __restrict__ bB,
                           const float* __restrict__ gamma, float* __restrict__ out,
                           float* __restrict__ csum) {
  size_t idx = (size_t)blockIdx.x * 256 + threadIdx.x;
  int c = (idx >> 14) & (CH - 1);                    // HWSZ = 2^14
  int b = idx >> 20;                                 // CH*HWSZ = 2^20
  float a = fmaxf(sB[b*CH + c] * pw[idx] + bB[b*CH + c], 0.f);
  float o = x[idx] + gamma[0] * a;
  out[idx] = o;
  __shared__ float red[256];
  red[threadIdx.x] = o;
  __syncthreads();
  for (int off = 128; off > 0; off >>= 1) {
    if (threadIdx.x < off) red[threadIdx.x] += red[threadIdx.x + off];
    __syncthreads();
  }
  if (threadIdx.x == 0) atomicAdd(&csum[b*CH + c], red[0]);
}

// ============ ECA gate: conv1d(k=3, pad=1) over channels + sigmoid ============
__global__ void k_eca(const float* __restrict__ csum, const float* __restrict__ eca_w,
                      float* __restrict__ gate) {
  int b = blockIdx.x, c = threadIdx.x;               // 64 threads
  __shared__ float m[CH];
  m[c] = csum[b*CH + c] * (1.0f / (float)HWSZ);
  __syncthreads();
  float left  = (c > 0)      ? m[c-1] : 0.f;
  float right = (c < CH - 1) ? m[c+1] : 0.f;
  float g = eca_w[0]*left + eca_w[1]*m[c] + eca_w[2]*right;
  gate[b*CH + c] = 1.f / (1.f + expf(-g));
}

// ============ apply channel gate in place ============
__global__ void k_scale(float* __restrict__ out, const float* __restrict__ gate) {
  size_t idx = (size_t)blockIdx.x * 256 + threadIdx.x;
  int c = (idx >> 14) & (CH - 1);
  int b = idx >> 20;
  out[idx] *= gate[b*CH + c];
}

// =====================================================================
extern "C" void kernel_launch(void* const* d_in, const int* in_sizes, int n_in,
                              void* d_out, int out_size, void* d_ws, size_t ws_size,
                              hipStream_t stream) {
  const float* x        = (const float*)d_in[0];
  const float* dw3      = (const float*)d_in[1];
  const float* dw5      = (const float*)d_in[2];
  const float* g1_w     = (const float*)d_in[3];
  const float* g1_gn_w  = (const float*)d_in[4];
  const float* g1_gn_b  = (const float*)d_in[5];
  const float* g2a_w    = (const float*)d_in[6];
  const float* g2b_w    = (const float*)d_in[7];
  const float* g2b_b    = (const float*)d_in[8];
  const float* post_gn_w= (const float*)d_in[9];
  const float* post_gn_b= (const float*)d_in[10];
  const float* pw_w     = (const float*)d_in[11];
  const float* pw_gn_w  = (const float*)d_in[12];
  const float* pw_gn_b  = (const float*)d_in[13];
  const float* eca_w    = (const float*)d_in[14];
  const float* gamma    = (const float*)d_in[15];
  float* out = (float*)d_out;

  // workspace layout (floats)
  float* ws = (float*)d_ws;
  const size_t N_FULL = (size_t)BATCH * CH * HWSZ;   // 16777216
  float* ws_y      = ws;                             // fused branch output
  float* ws_pw     = ws_y   + N_FULL;                // pointwise conv raw output
  float* ws_g1raw  = ws_pw  + N_FULL;                // g1 conv raw (reused for h)
  float* ws_g1cat  = ws_g1raw + (size_t)BATCH*CG*PL; // 34-ch gate input
  float* ws_mlow   = ws_g1cat + (size_t)BATCH*34*PL;
  float* ws_logits = ws_mlow  + (size_t)BATCH*PL;
  float* ws_sg1    = ws_logits + (size_t)BATCH*2*PL;
  float* ws_bg1    = ws_sg1 + BATCH*CG;
  float* ws_sA     = ws_bg1 + BATCH*CG;
  float* ws_bA     = ws_sA  + BATCH*CH;
  float* ws_sB     = ws_bA  + BATCH*CH;
  float* ws_bB     = ws_sB  + BATCH*CH;
  float* ws_csum   = ws_bB  + BATCH*CH;
  float* ws_gate   = ws_csum + BATCH*CH;
  float* ws_h      = ws_g1raw;                       // reuse g1raw region for h

  const int convBlocks = (BATCH * (PL/16)) / 8;      // 4096 waves / 8 per block = 512

  // ---- gate path (low res) ----
  k_mlow<<<(BATCH*PL)/256, 256, 0, stream>>>(x, ws_mlow);
  k_conv3x3_wmma<CH, HH, 2, false><<<convBlocks, 256, 0, stream>>>(x, g1_w, ws_g1raw);
  k_gnstats<<<BATCH*CG, 256, 0, stream>>>(ws_g1raw, g1_gn_w, g1_gn_b,
                                          ws_sg1, ws_bg1, 1, PL, CG);
  k_g1cat<<<(BATCH*34*PL)/256, 256, 0, stream>>>(ws_g1raw, ws_sg1, ws_bg1,
                                                 ws_mlow, ws_g1cat);
  k_conv3x3_wmma<34, HL, 1, true><<<convBlocks, 256, 0, stream>>>(ws_g1cat, g2a_w, ws_h);
  k_logits<<<(BATCH*PL)/256, 256, 0, stream>>>(ws_h, g2b_w, g2b_b, ws_logits);

  // ---- fused dual depthwise + softmax gate ----
  k_fuse<<<(unsigned)(N_FULL/256), 256, 0, stream>>>(x, dw3, dw5, ws_logits, ws_y);

  // ---- post-GN stats, then WMMA pointwise GEMM with fused GN+ReLU ----
  k_gnstats<<<BATCH*32, 256, 0, stream>>>(ws_y, post_gn_w, post_gn_b,
                                          ws_sA, ws_bA, 2, HWSZ, CH);
  k_pwgemm<<<2048, 256, 0, stream>>>(ws_y, pw_w, ws_sA, ws_bA, ws_pw);
  k_gnstats<<<BATCH*32, 256, 0, stream>>>(ws_pw, pw_gn_w, pw_gn_b,
                                          ws_sB, ws_bB, 2, HWSZ, CH);

  // ---- residual + ECA channel gate ----
  k_zero<<<4, 256, 0, stream>>>(ws_csum, BATCH*CH);
  k_residual<<<(unsigned)(N_FULL/256), 256, 0, stream>>>(x, ws_pw, ws_sB, ws_bB,
                                                         gamma, out, ws_csum);
  k_eca<<<BATCH, CH, 0, stream>>>(ws_csum, eca_w, ws_gate);
  k_scale<<<(unsigned)(N_FULL/256), 256, 0, stream>>>(out, ws_gate);
}
